// RBF_16655883173981
// MI455X (gfx1250) — matrix operand order repaired and checked by reading, not compiled, not run on hardware
//
#include <hip/hip_runtime.h>

// ---------------------------------------------------------------------------
// RBF pairwise squared distances on gfx1250 (MI455X):
//   out[n,o] = ||x_n||^2 + ||w_o||^2 - 2 * sum_k x[n,k]*w[o,k]
//
// Strategy: bf16x3 split (x = hi + lo bf16; x*w ~= hi*hi + hi*lo + lo*hi) run
// on v_wmma_f32_16x16x32_bf16 (~fp32 accuracy at bf16 matrix-core rates).
// The fp32->bf16 split + row norms are hoisted into a one-pass prep kernel
// (workspace), so the GEMM hot loop has ZERO conversion VALU: it is pure
// global_load_b128 -> ds_store_b128 -> ds_load_b128 -> v_wmma, software-
// pipelined one K-chunk ahead. Fallback (small workspace): fused variant that
// converts in-loop.
// ---------------------------------------------------------------------------

typedef __attribute__((ext_vector_type(4)))  float          f32x4;
typedef __attribute__((ext_vector_type(8)))  float          v8f;
typedef __attribute__((ext_vector_type(16))) __bf16         v16bf;
typedef __attribute__((ext_vector_type(4)))  unsigned short u16x4;
typedef __attribute__((ext_vector_type(8)))  unsigned short u16x8;

#define NROWS 16384   // samples (M)
#define KDIM  512     // in_features (K)
#define ODIM  4096    // out_features (N)
#define BM    128     // workgroup M tile
#define BN    128     // workgroup N tile
#define BK    32      // K chunk (one bf16 WMMA depth)
#define LDSK  40      // padded LDS row stride in shorts (80B: 16B-aligned, bank-spread)

union FragU { u16x8 h[2]; v16bf v; };

__device__ __forceinline__ unsigned short f32_to_bf16(float f) {
  unsigned u = __builtin_bit_cast(unsigned, f);
  u += 0x7FFFu + ((u >> 16) & 1u);              // round-to-nearest-even
  return (unsigned short)(u >> 16);
}
__device__ __forceinline__ float bf16_to_f32(unsigned short h) {
  unsigned u = ((unsigned)h) << 16;
  return __builtin_bit_cast(float, u);
}

#define WMMA_BF16(A, B, C) \
  __builtin_amdgcn_wmma_f32_16x16x32_bf16(false, (A), false, (B), (short)0, (C), false, false)

// ---------------------------------------------------------------------------
// Prep pass: one block per row. fp32 row -> hi/lo bf16 planes + ||row||^2.
// ---------------------------------------------------------------------------
__global__ __launch_bounds__(128)
void split_bf16x2_rows_kernel(const float* __restrict__ src,
                              unsigned short* __restrict__ hi,
                              unsigned short* __restrict__ lo,
                              float* __restrict__ norm) {
  const int row = blockIdx.x;
  const int t   = threadIdx.x;                  // 128 threads = 4 wave32
  const size_t off = (size_t)row * KDIM + t * 4;

  f32x4 v = *(const f32x4*)(src + off);
  u16x4 h4, l4;
  #pragma unroll
  for (int c = 0; c < 4; ++c) {
    unsigned short h = f32_to_bf16(v[c]);
    h4[c] = h;
    l4[c] = f32_to_bf16(v[c] - bf16_to_f32(h));
  }
  *(u16x4*)(hi + off) = h4;
  *(u16x4*)(lo + off) = l4;

  float s = v[0]*v[0] + v[1]*v[1] + v[2]*v[2] + v[3]*v[3];
  s += __shfl_xor(s, 1);  s += __shfl_xor(s, 2);  s += __shfl_xor(s, 4);
  s += __shfl_xor(s, 8);  s += __shfl_xor(s, 16);
  __shared__ float red[4];
  if ((t & 31) == 0) red[t >> 5] = s;
  __syncthreads();
  if (t == 0) norm[row] = red[0] + red[1] + red[2] + red[3];
}

// ---------------------------------------------------------------------------
// Main GEMM, pre-split inputs: hot loop has no conversions at all.
// 256 threads = 8 waves (2Mx4N); each wave owns 64x32 = 4x2 WMMA tiles.
// ---------------------------------------------------------------------------
__global__ __launch_bounds__(256)
void rbf_wmma_presplit_kernel(const unsigned short* __restrict__ Xhi,
                              const unsigned short* __restrict__ Xlo,
                              const unsigned short* __restrict__ Whi,
                              const unsigned short* __restrict__ Wlo,
                              const float* __restrict__ x2g,
                              const float* __restrict__ w2g,
                              float* __restrict__ out) {
  __shared__ __align__(16) unsigned short sAhi[BM * LDSK];
  __shared__ __align__(16) unsigned short sAlo[BM * LDSK];
  __shared__ __align__(16) unsigned short sBhi[BN * LDSK];
  __shared__ __align__(16) unsigned short sBlo[BN * LDSK];
  __shared__ float sX2[BM];
  __shared__ float sW2[BN];

  const int t    = threadIdx.x;
  const int lane = t & 31;
  const int wid  = t >> 5;

  const int blockN0 = blockIdx.x * BN;
  const int blockM0 = blockIdx.y * BM;

  const int moff = (wid >> 2) * 64;
  const int noff = (wid & 3)  * 32;

  // Fragment lane decomposition (contiguous k order used here for both A and B):
  //  A 16x32: lanes 0-15 hold K 0-15;        lanes 16-31 hold K 16-31
  //  B 32x16: lanes 0-15 hold K 0-15;        lanes 16-31 hold K 16-31
  const int flr = lane & 15;
  const int ka  = (lane >> 4) << 3;
  const int kb  = (lane >> 4) << 4;

  // Staging map: thread t handles rows {t>>2, t>>2+64}, short-cols (t&3)*8..+7
  const int srow = t >> 2;          // 0..63
  const int s8c  = (t & 3) * 8;     // 0,8,16,24 shorts

  if (t < BM) sX2[t] = x2g[blockM0 + t];
  if (t < BN) sW2[t] = w2g[blockN0 + t];

  v8f acc[4][2];
  #pragma unroll
  for (int i = 0; i < 4; ++i)
    #pragma unroll
    for (int j = 0; j < 2; ++j)
      acc[i][j] = (v8f){0.f,0.f,0.f,0.f,0.f,0.f,0.f,0.f};

  // Preload chunk 0 into registers (software pipeline depth 1)
  u16x8 cxh[2], cxl[2], cwh[2], cwl[2];
  #pragma unroll
  for (int i = 0; i < 2; ++i) {
    const size_t xr = (size_t)(blockM0 + srow + 64 * i) * KDIM + s8c;
    const size_t wr = (size_t)(blockN0 + srow + 64 * i) * KDIM + s8c;
    cxh[i] = *(const u16x8*)(Xhi + xr);
    cxl[i] = *(const u16x8*)(Xlo + xr);
    cwh[i] = *(const u16x8*)(Whi + wr);
    cwl[i] = *(const u16x8*)(Wlo + wr);
  }

  for (int kc = 0; kc < KDIM; kc += BK) {
    // commit staged registers to LDS
    #pragma unroll
    for (int i = 0; i < 2; ++i) {
      const int b = (srow + 64 * i) * LDSK + s8c;
      *(u16x8*)&sAhi[b] = cxh[i];
      *(u16x8*)&sAlo[b] = cxl[i];
      *(u16x8*)&sBhi[b] = cwh[i];
      *(u16x8*)&sBlo[b] = cwl[i];
    }
    __syncthreads();

    // issue next chunk's global loads; they overlap the WMMAs below
    if (kc + BK < KDIM) {
      #pragma unroll
      for (int i = 0; i < 2; ++i) {
        const size_t xr = (size_t)(blockM0 + srow + 64 * i) * KDIM + (kc + BK) + s8c;
        const size_t wr = (size_t)(blockN0 + srow + 64 * i) * KDIM + (kc + BK) + s8c;
        cxh[i] = *(const u16x8*)(Xhi + xr);
        cxl[i] = *(const u16x8*)(Xlo + xr);
        cwh[i] = *(const u16x8*)(Whi + wr);
        cwl[i] = *(const u16x8*)(Wlo + wr);
      }
    }

    // gather fragments from LDS
    FragU ahi[4], alo[4], bhi[2], blo[2];
    #pragma unroll
    for (int i = 0; i < 4; ++i) {
      const int row = (moff + i * 16 + flr) * LDSK;
      ahi[i].h[0] = *(const u16x8*)&sAhi[row + kb];
      ahi[i].h[1] = *(const u16x8*)&sAhi[row + kb + 8];
      alo[i].h[0] = *(const u16x8*)&sAlo[row + kb];
      alo[i].h[1] = *(const u16x8*)&sAlo[row + kb + 8];
    }
    #pragma unroll
    for (int j = 0; j < 2; ++j) {
      const int row = (noff + j * 16 + flr) * LDSK;
      bhi[j].h[0] = *(const u16x8*)&sBhi[row + kb];
      bhi[j].h[1] = *(const u16x8*)&sBhi[row + kb + 8];
      blo[j].h[0] = *(const u16x8*)&sBlo[row + kb];
      blo[j].h[1] = *(const u16x8*)&sBlo[row + kb + 8];
    }

    // bf16x3 matrix-core accumulation
    #pragma unroll
    for (int i = 0; i < 4; ++i)
      #pragma unroll
      for (int j = 0; j < 2; ++j) {
        acc[i][j] = WMMA_BF16(ahi[i].v, bhi[j].v, acc[i][j]);
        acc[i][j] = WMMA_BF16(ahi[i].v, blo[j].v, acc[i][j]);
        acc[i][j] = WMMA_BF16(alo[i].v, bhi[j].v, acc[i][j]);
      }
    __syncthreads();
  }

  // epilogue: out = x2 + w2 - 2*acc (16x16 f32 C-layout)
  const int rhalf = (lane >> 4) << 3;
  const int cl    = lane & 15;
  #pragma unroll
  for (int i = 0; i < 4; ++i) {
    #pragma unroll
    for (int j = 0; j < 2; ++j) {
      const int c     = noff + j * 16 + cl;
      const float w2v = sW2[c];
      const size_t gc = (size_t)(blockN0 + c);
      v8f a = acc[i][j];
      #pragma unroll
      for (int r = 0; r < 8; ++r) {
        const int lr = moff + i * 16 + rhalf + r;
        out[(size_t)(blockM0 + lr) * ODIM + gc] = sX2[lr] + w2v - 2.0f * a[r];
      }
    }
  }
}

// ---------------------------------------------------------------------------
// Fallback: fused variant (conversion in-loop), used if workspace too small.
// ---------------------------------------------------------------------------
__global__ __launch_bounds__(256)
void rbf_wmma_fused_kernel(const float* __restrict__ X,
                           const float* __restrict__ W,
                           float* __restrict__ out) {
  __shared__ __align__(16) unsigned short sAhi[BM * LDSK];
  __shared__ __align__(16) unsigned short sAlo[BM * LDSK];
  __shared__ __align__(16) unsigned short sBhi[BN * LDSK];
  __shared__ __align__(16) unsigned short sBlo[BN * LDSK];
  __shared__ float sX2[BM];
  __shared__ float sW2[BN];

  const int t    = threadIdx.x;
  const int lane = t & 31;
  const int wid  = t >> 5;
  const int blockN0 = blockIdx.x * BN;
  const int blockM0 = blockIdx.y * BM;
  const int moff = (wid >> 2) * 64;
  const int noff = (wid & 3)  * 32;
  const int rbase = t >> 3;
  const int c4    = t & 7;
  const int flr = lane & 15;
  const int ka  = (lane >> 4) << 3;
  const int kb  = (lane >> 4) << 4;

  v8f acc[4][2];
  #pragma unroll
  for (int i = 0; i < 4; ++i)
    #pragma unroll
    for (int j = 0; j < 2; ++j)
      acc[i][j] = (v8f){0.f,0.f,0.f,0.f,0.f,0.f,0.f,0.f};

  float xs[4] = {0.f,0.f,0.f,0.f};
  float ws[4] = {0.f,0.f,0.f,0.f};

  for (int kc = 0; kc < KDIM; kc += BK) {
    f32x4 xv[4], wv[4];
    #pragma unroll
    for (int i = 0; i < 4; ++i) {
      const int row = rbase + 32 * i;
      xv[i] = *(const f32x4*)(X + (size_t)(blockM0 + row) * KDIM + kc + c4 * 4);
      wv[i] = *(const f32x4*)(W + (size_t)(blockN0 + row) * KDIM + kc + c4 * 4);
    }
    #pragma unroll
    for (int i = 0; i < 4; ++i) {
      const int row  = rbase + 32 * i;
      const int base = row * LDSK + c4 * 4;
      f32x4 vx = xv[i], vw = wv[i];
      xs[i] += vx[0]*vx[0] + vx[1]*vx[1] + vx[2]*vx[2] + vx[3]*vx[3];
      ws[i] += vw[0]*vw[0] + vw[1]*vw[1] + vw[2]*vw[2] + vw[3]*vw[3];
      u16x4 xhi, xlo, whi, wlo;
      #pragma unroll
      for (int c = 0; c < 4; ++c) {
        unsigned short h;
        h = f32_to_bf16(vx[c]); xhi[c] = h; xlo[c] = f32_to_bf16(vx[c] - bf16_to_f32(h));
        h = f32_to_bf16(vw[c]); whi[c] = h; wlo[c] = f32_to_bf16(vw[c] - bf16_to_f32(h));
      }
      *(u16x4*)&sAhi[base] = xhi;
      *(u16x4*)&sAlo[base] = xlo;
      *(u16x4*)&sBhi[base] = whi;
      *(u16x4*)&sBlo[base] = wlo;
    }
    __syncthreads();

    FragU ahi[4], alo[4], bhi[2], blo[2];
    #pragma unroll
    for (int i = 0; i < 4; ++i) {
      const int row = (moff + i * 16 + flr) * LDSK;
      ahi[i].h[0] = *(const u16x8*)&sAhi[row + kb];
      ahi[i].h[1] = *(const u16x8*)&sAhi[row + kb + 8];
      alo[i].h[0] = *(const u16x8*)&sAlo[row + kb];
      alo[i].h[1] = *(const u16x8*)&sAlo[row + kb + 8];
    }
    #pragma unroll
    for (int j = 0; j < 2; ++j) {
      const int row = (noff + j * 16 + flr) * LDSK;
      bhi[j].h[0] = *(const u16x8*)&sBhi[row + kb];
      bhi[j].h[1] = *(const u16x8*)&sBhi[row + kb + 8];
      blo[j].h[0] = *(const u16x8*)&sBlo[row + kb];
      blo[j].h[1] = *(const u16x8*)&sBlo[row + kb + 8];
    }

    #pragma unroll
    for (int i = 0; i < 4; ++i)
      #pragma unroll
      for (int j = 0; j < 2; ++j) {
        acc[i][j] = WMMA_BF16(ahi[i].v, bhi[j].v, acc[i][j]);
        acc[i][j] = WMMA_BF16(ahi[i].v, blo[j].v, acc[i][j]);
        acc[i][j] = WMMA_BF16(alo[i].v, bhi[j].v, acc[i][j]);
      }
    __syncthreads();
  }

  #pragma unroll
  for (int i = 0; i < 4; ++i) {
    float vx = xs[i], vw = ws[i];
    vx += __shfl_xor(vx, 1); vx += __shfl_xor(vx, 2); vx += __shfl_xor(vx, 4);
    vw += __shfl_xor(vw, 1); vw += __shfl_xor(vw, 2); vw += __shfl_xor(vw, 4);
    if ((t & 7) == 0) {
      sX2[32 * i + (t >> 3)] = vx;
      sW2[32 * i + (t >> 3)] = vw;
    }
  }
  __syncthreads();

  const int rhalf = (lane >> 4) << 3;
  const int cl    = lane & 15;
  #pragma unroll
  for (int i = 0; i < 4; ++i) {
    #pragma unroll
    for (int j = 0; j < 2; ++j) {
      const int c     = noff + j * 16 + cl;
      const float w2v = sW2[c];
      const size_t gc = (size_t)(blockN0 + c);
      v8f a = acc[i][j];
      #pragma unroll
      for (int r = 0; r < 8; ++r) {
        const int lr = moff + i * 16 + rhalf + r;
        out[(size_t)(blockM0 + lr) * ODIM + gc] = sX2[lr] + w2v - 2.0f * a[r];
      }
    }
  }
}

extern "C" void kernel_launch(void* const* d_in, const int* in_sizes, int n_in,
                              void* d_out, int out_size, void* d_ws, size_t ws_size,
                              hipStream_t stream) {
  (void)in_sizes; (void)n_in; (void)out_size;
  const float* x = (const float*)d_in[0];   // (16384, 512) fp32
  const float* w = (const float*)d_in[1];   // (4096, 512) fp32
  float* out = (float*)d_out;               // (16384, 4096) fp32

  const size_t XE = (size_t)NROWS * KDIM;   // X elements
  const size_t WE = (size_t)ODIM  * KDIM;   // W elements
  const size_t need = 2 * XE * sizeof(unsigned short)   // Xhi, Xlo
                    + 2 * WE * sizeof(unsigned short)   // Whi, Wlo
                    + NROWS * sizeof(float)             // x2
                    + ODIM  * sizeof(float);            // w2

  dim3 grid(ODIM / BN, NROWS / BM);         // (32, 128)
  dim3 block(256);                          // 8 wave32s

  if (d_ws != nullptr && ws_size >= need) {
    unsigned short* Xhi = (unsigned short*)d_ws;
    unsigned short* Xlo = Xhi + XE;
    unsigned short* Whi = Xlo + XE;
    unsigned short* Wlo = Whi + WE;
    float* x2 = (float*)(Wlo + WE);
    float* w2 = x2 + NROWS;

    split_bf16x2_rows_kernel<<<NROWS, 128, 0, stream>>>(x, Xhi, Xlo, x2);
    split_bf16x2_rows_kernel<<<ODIM,  128, 0, stream>>>(w, Whi, Wlo, w2);
    rbf_wmma_presplit_kernel<<<grid, block, 0, stream>>>(Xhi, Xlo, Whi, Wlo, x2, w2, out);
  } else {
    rbf_wmma_fused_kernel<<<grid, block, 0, stream>>>(x, w, out);
  }
}
